// LstmMamba_2_56942676411239
// MI455X (gfx1250) — matrix operand order repaired and checked by reading, not compiled, not run on hardware
//
#include <hip/hip_runtime.h>
#include <hip/hip_bf16.h>

// ---------------- model dims ----------------
#define Vv   50280
#define Dd   768
#define NLn  24
#define NSn  16
#define Kk   4
#define DIi  1536
#define Rr   48
#define Hh   200
#define D2d  800
#define DI2d 1600
#define R2r  50
#define NLAB 3
#define Bb   2
#define Ll   512
#define Mm   (Bb*Ll)

typedef __bf16 v16bf __attribute__((ext_vector_type(16)));
typedef __bf16 v8bf  __attribute__((ext_vector_type(8)));
typedef float  v8f   __attribute__((ext_vector_type(8)));
typedef unsigned int u32x4 __attribute__((ext_vector_type(4)));
typedef int          i32x4 __attribute__((ext_vector_type(4)));
typedef int          i32x8 __attribute__((ext_vector_type(8)));
typedef int          v4i   __attribute__((ext_vector_type(4)));
typedef __attribute__((address_space(1))) v4i* gas1_v4i;   // global v4i*
typedef __attribute__((address_space(3))) v4i* las3_v4i;   // LDS v4i* (32-bit)

__device__ __forceinline__ float siluf(float x) { return x / (1.f + __expf(-x)); }
__device__ __forceinline__ float sigm(float x)  { return 1.f / (1.f + __expf(-x)); }
__device__ __forceinline__ float softplusf(float x) { return (x > 20.f) ? x : log1pf(__expf(x)); }

// ---- CDNA5 async global->LDS (ASYNCcnt) with sync fallback ----
__device__ __forceinline__ void async_ld_b128(const __bf16* g, __bf16* l) {
#if __has_builtin(__builtin_amdgcn_global_load_async_to_lds_b128)
  __builtin_amdgcn_global_load_async_to_lds_b128(
      (gas1_v4i)(size_t)g, (las3_v4i)(unsigned)(size_t)l, 0, 0);
#else
  *(uint4*)l = *(const uint4*)g;
#endif
}
__device__ __forceinline__ void wait_async0() {
#if __has_builtin(__builtin_amdgcn_s_wait_asynccnt)
  __builtin_amdgcn_s_wait_asynccnt(0);
#elif __has_builtin(__builtin_amdgcn_global_load_async_to_lds_b128)
  asm volatile("s_wait_asynccnt 0x0" ::: "memory");
#endif
}
__device__ __forceinline__ void wait_tensor0() {
#if __has_builtin(__builtin_amdgcn_s_wait_tensorcnt) && __has_builtin(__builtin_amdgcn_tensor_load_to_lds)
  __builtin_amdgcn_s_wait_tensorcnt(0);
#endif
}

__device__ __forceinline__ v16bf cat16(v8bf a, v8bf b) {
  return __builtin_shufflevector(a, b, 0,1,2,3,4,5,6,7,8,9,10,11,12,13,14,15);
}

// ---------------- f32 -> bf16 convert ----------------
__global__ void k_cvt_bf16(const float* __restrict__ s, __bf16* __restrict__ d, int n) {
  int i = blockIdx.x * 256 + threadIdx.x;
  if (i < n) d[i] = (__bf16)s[i];
}

// ---------------- embedding gather ----------------
__global__ __launch_bounds__(256)
void k_embed(const int* __restrict__ ids, const float* __restrict__ emb, float* __restrict__ h) {
  int m = blockIdx.x;
  int tok = ids[m];
  for (int d = threadIdx.x; d < Dd; d += 256)
    h[(size_t)m * Dd + d] = emb[(size_t)tok * Dd + d];
}

// ---------------- rmsnorm ----------------
__global__ __launch_bounds__(256)
void k_rmsnorm(const float* __restrict__ x, const float* __restrict__ w,
               __bf16* __restrict__ ob, float* __restrict__ of) {
  __shared__ float red[8];
  int m = blockIdx.x;
  const float* row = x + (size_t)m * Dd;
  float s = 0.f;
  for (int i = threadIdx.x; i < Dd; i += 256) { float v = row[i]; s += v * v; }
  for (int o = 16; o > 0; o >>= 1) s += __shfl_down(s, o, 32);
  if ((threadIdx.x & 31) == 0) red[threadIdx.x >> 5] = s;
  __syncthreads();
  if (threadIdx.x == 0) {
    float t = 0.f;
    for (int i = 0; i < 8; ++i) t += red[i];
    red[0] = rsqrtf(t / (float)Dd + 1e-5f);
  }
  __syncthreads();
  float sc = red[0];
  for (int i = threadIdx.x; i < Dd; i += 256) {
    float v = row[i] * sc * w[i];
    if (ob) ob[(size_t)m * Dd + i] = (__bf16)v;
    else    of[(size_t)m * Dd + i] = v;
  }
}

// =====================================================================
// bf16 WMMA GEMM:  C[M,N] = (resid?) + A[M,K] * W[N,K]^T
// 256 threads = 8 waves (wave32), block tile 128x64, wave tile 32x32
// (2x2 of 16x16 -> 4 v_wmma per K-step), BK=32, double-buffered LDS.
// A tile staged with GLOBAL_LOAD_ASYNC_TO_LDS (ASYNCcnt); B (weight)
// tile staged with the Tensor Data Mover (TENSORcnt), LDS-padded to an
// 80B row stride (pad_interval=16DW, pad_amount=4DW) for conflict-free
// ds_load_b128 fragment reads. TDM zero-fills rows beyond N (xproj N=80).
// =====================================================================
#define BM 128
#define BN 64
#define BK 32
#define ASTR 40   // bf16 elems per LDS row (64B data + 16B pad)

__global__ __launch_bounds__(256)
void k_gemm_bf16(float* __restrict__ C, const __bf16* __restrict__ A,
                 const __bf16* __restrict__ W, const float* __restrict__ resid,
                 int M, int N, int K) {
  __shared__ __bf16 smA[2][BM][ASTR];
  __shared__ __bf16 smB[2][BN][ASTR];
  int tid  = threadIdx.x;
  int bm   = blockIdx.x * BM;
  int bn   = blockIdx.y * BN;
  int wave = tid >> 5, lane = tid & 31;
  int wm = wave & 3, wn = wave >> 2;       // 4 x 2 wave grid
  int half = lane >> 4, l16 = lane & 15;

  v8f acc[2][2] = {};

  int arow = tid >> 1;                     // 0..127
  int acol = (tid & 1) * 16;               // 0 or 16

  auto stageA = [&](int buf, int kk) {
    const __bf16* g = A + (size_t)(bm + arow) * K + kk + acol;
    async_ld_b128(g,     &smA[buf][arow][acol]);
    async_ld_b128(g + 8, &smA[buf][arow][acol + 8]);
  };

  auto stageB = [&](int buf, int kk) {
#if __has_builtin(__builtin_amdgcn_tensor_load_to_lds)
    if (wave == 0) {
      unsigned lds_off = (unsigned)(size_t)&smB[buf][0][0];
      unsigned long long ga =
          (unsigned long long)(size_t)W + (((size_t)bn * K + kk) << 1);
      u32x4 g0;
      g0.x = 1u;                                   // count=1 (valid D#)
      g0.y = lds_off;                              // LDS byte address
      g0.z = (unsigned)(ga & 0xffffffffull);       // global_addr[31:0]
      g0.w = (unsigned)((ga >> 32) & 0x1ffffffull) | (2u << 30);  // [56:32] | type=2
      unsigned td0 = (unsigned)(K - kk);           // remaining cols
      unsigned td1 = (unsigned)(N - bn);           // remaining rows (OOB -> 0-fill)
      unsigned long long st0 = (unsigned long long)K;
      i32x8 g1;
      g1[0] = (1 << 16)        // data_size = 2 bytes
            | (1 << 20)        // pad_enable
            | (3 << 22)        // pad_interval: 16 DWORDs
            | (3 << 25);       // pad_amount: 4 DWORDs -> 80B row stride
      g1[1] = (int)((td0 & 0xffffu) << 16);
      g1[2] = (int)((td0 >> 16) | ((td1 & 0xffffu) << 16));
      g1[3] = (int)((td1 >> 16) | (BK << 16));     // tile_dim0 = 32
      g1[4] = BN;                                  // tile_dim1 = 64 rows
      g1[5] = (int)(st0 & 0xffffffffull);          // tensor_dim0_stride lo
      g1[6] = (int)(st0 >> 32);                    // stride hi
      g1[7] = 0;
      i32x4 z4 = {0, 0, 0, 0};
#if __clang_major__ >= 23
      i32x8 z8 = {0, 0, 0, 0, 0, 0, 0, 0};
      __builtin_amdgcn_tensor_load_to_lds(g0, g1, z4, z4, z8, 0);
#else
      __builtin_amdgcn_tensor_load_to_lds(g0, g1, z4, z4, 0);
#endif
    }
#else
    int r = tid >> 2, c = (tid & 3) * 8;
    int nr = bn + r; if (nr >= N) nr = N - 1;
    *(uint4*)&smB[buf][r][c] = *(const uint4*)(W + (size_t)nr * K + kk + c);
#endif
  };

  stageA(0, 0);
  stageB(0, 0);
  wait_async0();
  wait_tensor0();
  __syncthreads();

  for (int kk = 0; kk < K; kk += BK) {
    int cur = (kk >> 5) & 1;
    if (kk + BK < K) { stageA(cur ^ 1, kk + BK); stageB(cur ^ 1, kk + BK); }

    // fragments (ISA 16-bit layouts): A lane<16 K{0..7,16..23}, lane>=16 K{8..15,24..31}
    const __bf16* pa0 = &smA[cur][wm * 32 + l16][0];
    const __bf16* pa1 = &smA[cur][wm * 32 + 16 + l16][0];
    v16bf a0 = cat16(*(const v8bf*)(pa0 + half * 8), *(const v8bf*)(pa0 + 16 + half * 8));
    v16bf a1 = cat16(*(const v8bf*)(pa1 + half * 8), *(const v8bf*)(pa1 + 16 + half * 8));
    const __bf16* pb0 = &smB[cur][wn * 32 + l16][0];
    const __bf16* pb1 = &smB[cur][wn * 32 + 16 + l16][0];
    v16bf b0 = cat16(*(const v8bf*)(pb0 + half * 8), *(const v8bf*)(pb0 + 16 + half * 8));
    v16bf b1 = cat16(*(const v8bf*)(pb1 + half * 8), *(const v8bf*)(pb1 + 16 + half * 8));

    acc[0][0] = __builtin_amdgcn_wmma_f32_16x16x32_bf16(false, a0, false, b0, (short)0, acc[0][0], false, false);
    acc[0][1] = __builtin_amdgcn_wmma_f32_16x16x32_bf16(false, a0, false, b1, (short)0, acc[0][1], false, false);
    acc[1][0] = __builtin_amdgcn_wmma_f32_16x16x32_bf16(false, a1, false, b0, (short)0, acc[1][0], false, false);
    acc[1][1] = __builtin_amdgcn_wmma_f32_16x16x32_bf16(false, a1, false, b1, (short)0, acc[1][1], false, false);

    wait_async0();
    wait_tensor0();
    __syncthreads();
  }

  // C layout: VGPR r -> M=r (lanes 0-15) / M=8+r (lanes 16-31); N = lane%16
#pragma unroll
  for (int i = 0; i < 2; ++i) {
#pragma unroll
    for (int j = 0; j < 2; ++j) {
      int n  = bn + wn * 32 + j * 16 + l16;
      int mb = bm + wm * 32 + i * 16 + half * 8;
      if (n < N) {
#pragma unroll
        for (int r = 0; r < 8; ++r) {
          size_t off = (size_t)(mb + r) * N + n;
          float v = acc[i][j][r];
          if (resid) v += resid[off];
          C[off] = v;
        }
      }
    }
  }
}

// ---------------- causal depthwise conv (K=4) + SiLU ----------------
__global__ __launch_bounds__(256)
void k_conv_silu(const float* __restrict__ xz, const float* __restrict__ cw,
                 const float* __restrict__ cb, float* __restrict__ u,
                 __bf16* __restrict__ ub) {
  int idx = blockIdx.x * 256 + threadIdx.x;
  if (idx >= Mm * DIi) return;
  int d = idx % DIi, m = idx / DIi;
  int l = m % Ll;
  float acc = cb[d];
  for (int k = 0; k < Kk; ++k) {
    int li = l + k - (Kk - 1);
    if (li >= 0) acc += xz[(size_t)(m + k - (Kk - 1)) * (2 * DIi) + d] * cw[d * Kk + k];
  }
  float s = siluf(acc);
  u[idx]  = s;
  ub[idx] = (__bf16)s;
}

// ---------------- dt = softplus(dbl[:, :R] @ dtw^T + dtb) ----------------
__global__ __launch_bounds__(256)
void k_dt(const float* __restrict__ dbl, const float* __restrict__ dtw,
          const float* __restrict__ dtb, float* __restrict__ dt) {
  __shared__ float r48[Rr];
  int m = blockIdx.x;
  if (threadIdx.x < Rr) r48[threadIdx.x] = dbl[(size_t)m * (Rr + 2 * NSn) + threadIdx.x];
  __syncthreads();
  for (int d = threadIdx.x; d < DIi; d += 256) {
    float acc = dtb[d];
    const float* wr = dtw + (size_t)d * Rr;
    for (int r = 0; r < Rr; ++r) acc += r48[r] * wr[r];
    dt[(size_t)m * DIi + d] = softplusf(acc);
  }
}

// ---------------- selective scan; fused y = (h.C + D*u)*silu(z) ----------------
__global__ __launch_bounds__(256)
void k_scan(const float* __restrict__ dt, const float* __restrict__ dbl,
            const float* __restrict__ u, const float* __restrict__ xz,
            const float* __restrict__ A_log, const float* __restrict__ Dp,
            __bf16* __restrict__ yb) {
  __shared__ float Bs[NSn], Cs[NSn];
  int b = blockIdx.x / (DIi / 256);
  int d = (blockIdx.x % (DIi / 256)) * 256 + threadIdx.x;
  float Aa[NSn], h[NSn];
  for (int n = 0; n < NSn; ++n) { Aa[n] = -__expf(A_log[d * NSn + n]); h[n] = 0.f; }
  float Dv = Dp[d];
  for (int l = 0; l < Ll; ++l) {
    int m = b * Ll + l;
    __syncthreads();
    if (threadIdx.x < 2 * NSn) {
      float v = dbl[(size_t)m * (Rr + 2 * NSn) + Rr + threadIdx.x];
      if (threadIdx.x < NSn) Bs[threadIdx.x] = v; else Cs[threadIdx.x - NSn] = v;
    }
    __syncthreads();
    float dtv = dt[(size_t)m * DIi + d];
    float uv  = u[(size_t)m * DIi + d];
    float acc = 0.f;
    for (int n = 0; n < NSn; ++n) {
      h[n] = __expf(dtv * Aa[n]) * h[n] + dtv * Bs[n] * uv;
      acc += h[n] * Cs[n];
    }
    float z = xz[(size_t)m * (2 * DIi) + DIi + d];
    yb[(size_t)m * DIi + d] = (__bf16)((acc + Dv * uv) * siluf(z));
  }
}

// ---------------- masked mean pool ----------------
__global__ void k_pool(const float* __restrict__ hn, const int* __restrict__ mask,
                       float* __restrict__ pooled) {
  int b = blockIdx.y;
  int d = blockIdx.x * 256 + threadIdx.x;
  if (d >= Dd) return;
  float acc = 0.f, ms = 0.f;
  for (int l = 0; l < Ll; ++l) {
    float mk = (float)mask[b * Ll + l];
    acc += hn[(size_t)(b * Ll + l) * Dd + d] * mk;
    ms  += mk;
  }
  pooled[b * Dd + d] = acc / ms;
}

// ---------------- LSTM cell ----------------
__global__ void k_lstm(const float* __restrict__ x, int xdim,
                       const float* __restrict__ h0, const float* __restrict__ c0,
                       const float* __restrict__ Wih, const float* __restrict__ Whh,
                       const float* __restrict__ bih, const float* __restrict__ bhh,
                       float* __restrict__ hOut, float* __restrict__ cOut) {
  int idx = blockIdx.x * 256 + threadIdx.x;
  if (idx >= Bb * Hh) return;
  int b = idx / Hh, j = idx % Hh;
  float g[4];
  for (int gi = 0; gi < 4; ++gi) {
    int row = gi * Hh + j;
    float acc = bih[row] + bhh[row];
    const float* wi = Wih + (size_t)row * xdim;
    const float* xr = x + (size_t)b * xdim;
    for (int k = 0; k < xdim; ++k) acc += xr[k] * wi[k];
    if (h0) {
      const float* wh = Whh + (size_t)row * Hh;
      const float* hr = h0 + b * Hh;
      for (int k = 0; k < Hh; ++k) acc += hr[k] * wh[k];
    }
    g[gi] = acc;
  }
  float cprev = c0 ? c0[idx] : 0.f;
  float c = sigm(g[1]) * cprev + sigm(g[0]) * tanhf(g[2]);
  hOut[idx] = sigm(g[3]) * tanhf(c);
  cOut[idx] = c;
}

// ---------------- pack [h1f, hf, h1b, hb] -> (B, 4H) ----------------
__global__ void k_pack(const float* __restrict__ h1f, const float* __restrict__ hf,
                       const float* __restrict__ h1b, const float* __restrict__ hb,
                       float* __restrict__ p2) {
  int idx = blockIdx.x * 256 + threadIdx.x;
  if (idx >= Bb * 4 * Hh) return;
  int b = idx / (4 * Hh), j = idx % (4 * Hh);
  const float* s = (j < Hh) ? h1f : (j < 2 * Hh) ? hf : (j < 3 * Hh) ? h1b : hb;
  p2[idx] = s[b * Hh + (j % Hh)];
}

// ---------------- generic small matvec ----------------
__global__ void k_matvec(float* __restrict__ out, int ldo,
                         const float* __restrict__ A, int lda,
                         const float* __restrict__ W, const float* __restrict__ bias,
                         int M, int N, int K, int act) {
  int idx = blockIdx.x * 256 + threadIdx.x;
  if (idx >= M * N) return;
  int b = idx / N, n = idx % N;
  float acc = bias ? bias[n] : 0.f;
  const float* ar = A + (size_t)b * lda;
  const float* wr = W + (size_t)n * K;
  for (int k = 0; k < K; ++k) acc += ar[k] * wr[k];
  if (act == 1) acc = softplusf(acc);
  out[(size_t)b * ldo + n] = acc;
}

// ---------------- L=1 mamba pieces ----------------
__global__ void k_conv1_silu(const float* __restrict__ xz2, const float* __restrict__ cw,
                             const float* __restrict__ cb, float* __restrict__ u2) {
  int idx = blockIdx.x * 256 + threadIdx.x;
  if (idx >= Bb * DI2d) return;
  int b = idx / DI2d, d = idx % DI2d;
  float a = cb[d] + xz2[(size_t)b * (2 * DI2d) + d] * cw[d * Kk + (Kk - 1)];
  u2[idx] = siluf(a);
}

__global__ void k_scan1(const float* __restrict__ dt2, const float* __restrict__ dbl2,
                        const float* __restrict__ u2, const float* __restrict__ xz2,
                        const float* __restrict__ Dp, float* __restrict__ y2) {
  int idx = blockIdx.x * 256 + threadIdx.x;
  if (idx >= Bb * DI2d) return;
  int b = idx / DI2d, d = idx % DI2d;
  float dtv = dt2[idx], uv = u2[idx], acc = 0.f;
  for (int n = 0; n < NSn; ++n) {
    float hh = dtv * dbl2[b * (R2r + 2 * NSn) + R2r + n] * uv;
    acc += hh * dbl2[b * (R2r + 2 * NSn) + R2r + NSn + n];
  }
  float z = xz2[(size_t)b * (2 * DI2d) + DI2d + d];
  y2[idx] = (acc + Dp[d] * uv) * siluf(z);
}

// ---------------- host driver ----------------
static inline size_t alignup(size_t x) { return (x + 255) & ~(size_t)255; }

extern "C" void kernel_launch(void* const* d_in, const int* in_sizes, int n_in,
                              void* d_out, int out_size, void* d_ws, size_t ws_size,
                              hipStream_t stream) {
  (void)in_sizes; (void)n_in; (void)out_size; (void)ws_size;
  const int*   ids    = (const int*)  d_in[0];
  const int*   mask   = (const int*)  d_in[1];
  const float* embed  = (const float*)d_in[2];
  const float* nw     = (const float*)d_in[3];
  const float* in_w   = (const float*)d_in[4];
  const float* conv_w = (const float*)d_in[5];
  const float* conv_b = (const float*)d_in[6];
  const float* xp_w   = (const float*)d_in[7];
  const float* dt_w   = (const float*)d_in[8];
  const float* dt_b   = (const float*)d_in[9];
  const float* A_log  = (const float*)d_in[10];
  const float* Dp     = (const float*)d_in[11];
  const float* out_w  = (const float*)d_in[12];
  const float* normf  = (const float*)d_in[13];
  const float* l1Wih  = (const float*)d_in[14];
  const float* l1Whh  = (const float*)d_in[15];
  const float* l1bih  = (const float*)d_in[16];
  const float* l1bhh  = (const float*)d_in[17];
  const float* l2Wih  = (const float*)d_in[18];
  const float* l2Whh  = (const float*)d_in[19];
  const float* l2bih  = (const float*)d_in[20];
  const float* l2bhh  = (const float*)d_in[21];
  const float* fc_w   = (const float*)d_in[40];
  const float* fc_b   = (const float*)d_in[41];

  char* ws = (char*)d_ws;
  size_t off = 0;
  auto grab = [&](size_t bytes) { char* p = ws + off; off = alignup(off + bytes); return p; };
  float*  h    = (float*) grab((size_t)Mm * Dd * 4);
  __bf16* xn   = (__bf16*)grab((size_t)Mm * Dd * 2);
  float*  xz   = (float*) grab((size_t)Mm * 2 * DIi * 4);
  float*  u    = (float*) grab((size_t)Mm * DIi * 4);
  __bf16* ub   = (__bf16*)grab((size_t)Mm * DIi * 2);
  float*  dbl  = (float*) grab((size_t)Mm * (Rr + 2 * NSn) * 4);
  float*  dt   = (float*) grab((size_t)Mm * DIi * 4);
  __bf16* yb   = (__bf16*)grab((size_t)Mm * DIi * 2);
  __bf16* wb   = (__bf16*)grab((size_t)2 * DIi * Dd * 2);
  float*  hn   = (float*) grab((size_t)Mm * Dd * 4);
  float*  pool = (float*) grab((size_t)Bb * Dd * 4);
  float*  h1f  = (float*) grab(Bb * Hh * 4);
  float*  h1b  = (float*) grab(Bb * Hh * 4);
  float*  hf   = (float*) grab(Bb * Hh * 4);
  float*  cf   = (float*) grab(Bb * Hh * 4);
  float*  hbk  = (float*) grab(Bb * Hh * 4);
  float*  cbk  = (float*) grab(Bb * Hh * 4);
  float*  junk = (float*) grab(Bb * Hh * 4);
  float*  p2   = (float*) grab(Bb * 4 * Hh * 4);
  float*  xz2  = (float*) grab(Bb * 2 * DI2d * 4);
  float*  u2   = (float*) grab(Bb * DI2d * 4);
  float*  dbl2 = (float*) grab(Bb * (R2r + 2 * NSn) * 4);
  float*  dt2  = (float*) grab(Bb * DI2d * 4);
  float*  y2   = (float*) grab(Bb * DI2d * 4);
  float*  x3   = (float*) grab(Bb * D2d * 4);
  float*  x4   = (float*) grab(Bb * D2d * 4);

  k_embed<<<Mm, 256, 0, stream>>>(ids, embed, h);

  for (int l = 0; l < NLn; ++l) {
    const float* w_in = in_w  + (size_t)l * 2 * DIi * Dd;
    const float* w_xp = xp_w  + (size_t)l * (Rr + 2 * NSn) * DIi;
    const float* w_ot = out_w + (size_t)l * Dd * DIi;

    k_rmsnorm<<<Mm, 256, 0, stream>>>(h, nw + (size_t)l * Dd, xn, nullptr);

    {   // in_proj: xz = xn @ in_w^T   (N=3072, K=768)
      int n = 2 * DIi * Dd;
      k_cvt_bf16<<<(n + 255) / 256, 256, 0, stream>>>(w_in, wb, n);
      dim3 g(Mm / BM, (2 * DIi + BN - 1) / BN);
      k_gemm_bf16<<<g, 256, 0, stream>>>(xz, xn, wb, nullptr, Mm, 2 * DIi, Dd);
    }

    k_conv_silu<<<(Mm * DIi + 255) / 256, 256, 0, stream>>>(
        xz, conv_w + (size_t)l * DIi * Kk, conv_b + (size_t)l * DIi, u, ub);

    {   // xproj: dbl = u @ xproj^T   (N=80, K=1536) — TDM zero-fills ragged rows
      int n = (Rr + 2 * NSn) * DIi;
      k_cvt_bf16<<<(n + 255) / 256, 256, 0, stream>>>(w_xp, wb, n);
      dim3 g(Mm / BM, ((Rr + 2 * NSn) + BN - 1) / BN);
      k_gemm_bf16<<<g, 256, 0, stream>>>(dbl, ub, wb, nullptr, Mm, Rr + 2 * NSn, DIi);
    }

    k_dt<<<Mm, 256, 0, stream>>>(dbl, dt_w + (size_t)l * DIi * Rr,
                                 dt_b + (size_t)l * DIi, dt);

    k_scan<<<Bb * (DIi / 256), 256, 0, stream>>>(
        dt, dbl, u, xz, A_log + (size_t)l * DIi * NSn, Dp + (size_t)l * DIi, yb);

    {   // out_proj + residual: h = h + y @ out_w^T   (N=768, K=1536)
      int n = Dd * DIi;
      k_cvt_bf16<<<(n + 255) / 256, 256, 0, stream>>>(w_ot, wb, n);
      dim3 g(Mm / BM, (Dd + BN - 1) / BN);
      k_gemm_bf16<<<g, 256, 0, stream>>>(h, yb, wb, h, Mm, Dd, DIi);
    }
  }

  k_rmsnorm<<<Mm, 256, 0, stream>>>(h, normf, nullptr, hn);
  { dim3 g((Dd + 255) / 256, Bb); k_pool<<<g, 256, 0, stream>>>(hn, mask, pool); }

  int lt = (Bb * Hh + 255) / 256;
  k_lstm<<<lt, 256, 0, stream>>>(pool, Dd, nullptr, nullptr, l1Wih, l1Whh, l1bih, l1bhh, h1f, junk);
  k_lstm<<<lt, 256, 0, stream>>>(pool, Dd, nullptr, nullptr, l1Wih + 4 * Hh * Dd,
                                 l1Whh + 4 * Hh * Hh, l1bih + 4 * Hh, l1bhh + 4 * Hh, h1b, junk);
  k_lstm<<<lt, 256, 0, stream>>>(h1f, Hh, nullptr, nullptr, l2Wih, l2Whh, l2bih, l2bhh, hf, cf);
  k_lstm<<<lt, 256, 0, stream>>>(h1b, Hh, hf, cf, l2Wih, l2Whh, l2bih, l2bhh, hf, cf);
  k_lstm<<<lt, 256, 0, stream>>>(h1b, Hh, nullptr, nullptr, l2Wih + 4 * Hh * Hh,
                                 l2Whh + 4 * Hh * Hh, l2bih + 4 * Hh, l2bhh + 4 * Hh, hbk, cbk);
  k_lstm<<<lt, 256, 0, stream>>>(h1f, Hh, hbk, cbk, l2Wih + 4 * Hh * Hh,
                                 l2Whh + 4 * Hh * Hh, l2bih + 4 * Hh, l2bhh + 4 * Hh, hbk, cbk);
  k_pack<<<(Bb * 4 * Hh + 255) / 256, 256, 0, stream>>>(h1f, hf, h1b, hbk, p2);

  const float* xin = p2;
  float* xout = x3;
  for (int p = 0; p < 2; ++p) {
    const float* mi = (const float*)d_in[22 + p * 9 + 0];
    const float* mc = (const float*)d_in[22 + p * 9 + 1];
    const float* mb = (const float*)d_in[22 + p * 9 + 2];
    const float* mx = (const float*)d_in[22 + p * 9 + 3];
    const float* md = (const float*)d_in[22 + p * 9 + 4];
    const float* mdb= (const float*)d_in[22 + p * 9 + 5];
    const float* mD = (const float*)d_in[22 + p * 9 + 7];
    const float* mo = (const float*)d_in[22 + p * 9 + 8];
    int NC = R2r + 2 * NSn;
    k_matvec<<<(Bb * 2 * DI2d + 255) / 256, 256, 0, stream>>>(
        xz2, 2 * DI2d, xin, D2d, mi, nullptr, Bb, 2 * DI2d, D2d, 0);
    k_conv1_silu<<<(Bb * DI2d + 255) / 256, 256, 0, stream>>>(xz2, mc, mb, u2);
    k_matvec<<<(Bb * NC + 255) / 256, 256, 0, stream>>>(
        dbl2, NC, u2, DI2d, mx, nullptr, Bb, NC, DI2d, 0);
    k_matvec<<<(Bb * DI2d + 255) / 256, 256, 0, stream>>>(
        dt2, DI2d, dbl2, NC, md, mdb, Bb, DI2d, R2r, 1);
    k_scan1<<<(Bb * DI2d + 255) / 256, 256, 0, stream>>>(dt2, dbl2, u2, xz2, mD, y2);
    k_matvec<<<(Bb * D2d + 255) / 256, 256, 0, stream>>>(
        xout, D2d, y2, DI2d, mo, nullptr, Bb, D2d, DI2d, 0);
    xin = xout; xout = x4;
  }

  k_matvec<<<1, 256, 0, stream>>>((float*)d_out, NLAB, x3, D2d, fc_w, fc_b,
                                  Bb, NLAB, D2d, 0);
}
